// MultiHeadAttention_8263517077672
// MI455X (gfx1250) — matrix operand order repaired and checked by reading, not compiled, not run on hardware
//
#include <hip/hip_runtime.h>
#include <hip/hip_bf16.h>

// ---------------------------------------------------------------------------
// MHA forward for MI455X (gfx1250, wave32).  ~60 GFLOP vs ~60MB traffic ->
// compute-bound -> bf16 WMMA path (v_wmma_f32_16x16x32_bf16, K-depth 8x the
// fp32 WMMA).  Pipeline:
//   k0a: x  (f32)  -> xb   (bf16)                 [one-time convert]
//   k0b: W1 (f32)  -> W1Tb (bf16, transposed)     [one-time convert+transpose]
//   k1 : QKV GEMM (M=4096,N=3072,K=1024), async double-buffered LDS staging,
//        epilogue de-interleaves q / kT / v into workspace (bf16)
//   k2 : flash attention, K/V blocks async-staged to LDS once per WG,
//        16 queries/wave, 32-key blocks, online softmax, f32 out
// ---------------------------------------------------------------------------

typedef __attribute__((ext_vector_type(16))) __bf16          v16bf;
typedef __attribute__((ext_vector_type(8)))  float           v8f;
typedef __attribute__((ext_vector_type(8)))  unsigned short  ush8;
typedef __attribute__((ext_vector_type(16))) unsigned short  ush16;
typedef __attribute__((ext_vector_type(4)))  float           f32x4;
typedef __attribute__((ext_vector_type(4)))  int             i32x4;

__device__ __forceinline__ unsigned short f2bf(float f) {
  union { float f; unsigned int u; } c; c.f = f;
  unsigned int u = c.u;
  u += 0x7FFFu + ((u >> 16) & 1u);          // round-to-nearest-even
  return (unsigned short)(u >> 16);
}

__device__ __forceinline__ v16bf mk16(ush8 a, ush8 b) {
  ush16 u = __builtin_shufflevector(a, b, 0,1,2,3,4,5,6,7,8,9,10,11,12,13,14,15);
  return __builtin_bit_cast(v16bf, u);
}

#define WMMA_BF16(A, B, C) \
  __builtin_amdgcn_wmma_f32_16x16x32_bf16(false, (A), false, (B), (short)0, (C), false, false)

// ---- gfx1250 async global->LDS copy (ASYNCcnt), with portable fallback ----
#if __has_builtin(__builtin_amdgcn_global_load_async_to_lds_b128) && \
    __has_builtin(__builtin_amdgcn_s_wait_asynccnt)
#define HAVE_ASYNC_LDS 1
#else
#define HAVE_ASYNC_LDS 0
#endif

__device__ __forceinline__ void cp16(const unsigned short* g, unsigned short* l) {
#if HAVE_ASYNC_LDS
  __builtin_amdgcn_global_load_async_to_lds_b128(
      (__attribute__((address_space(1))) i32x4*)g,
      (__attribute__((address_space(3))) i32x4*)l, 0, 0);
#else
  *(ush8*)l = *(const ush8*)g;
#endif
}
__device__ __forceinline__ void wait_async_grp() {   // allow newest group of 4
#if HAVE_ASYNC_LDS
  __builtin_amdgcn_s_wait_asynccnt(4);
#endif
}
__device__ __forceinline__ void wait_async_all() {
#if HAVE_ASYNC_LDS
  __builtin_amdgcn_s_wait_asynccnt(0);
#endif
}

// ---------------------------------------------------------------------------
// k0a: convert x [4096*1024] f32 -> bf16
// ---------------------------------------------------------------------------
__global__ __launch_bounds__(256) void convert_x_kernel(
    const float* __restrict__ x, unsigned short* __restrict__ xb)
{
  const size_t base = ((size_t)blockIdx.x * 256 + threadIdx.x) * 16;
  const f32x4* s4 = (const f32x4*)(x + base);
  f32x4 f0 = s4[0], f1 = s4[1], f2 = s4[2], f3 = s4[3];
  ush8 p0, p1;
  #pragma unroll
  for (int i = 0; i < 4; ++i) {
    p0[i]     = f2bf(f0[i]);  p0[4 + i] = f2bf(f1[i]);
    p1[i]     = f2bf(f2[i]);  p1[4 + i] = f2bf(f3[i]);
  }
  *(ush8*)&xb[base]     = p0;
  *(ush8*)&xb[base + 8] = p1;
}

// ---------------------------------------------------------------------------
// k0b: W1 [3072,1024] f32 -> W1Tb [1024,3072] bf16 (LDS-tiled transpose)
// ---------------------------------------------------------------------------
__global__ __launch_bounds__(256) void convert_w1t_kernel(
    const float* __restrict__ W1, unsigned short* __restrict__ W1Tb)
{
  __shared__ alignas(16) unsigned short Tld[64 * 80];   // padded rows
  const int tid = threadIdx.x;
  const int N0  = blockIdx.y * 64;     // over 3072
  const int K0  = blockIdx.x * 64;     // over 1024
  {
    const int nr = tid >> 2;           // 0..63
    const int kc = (tid & 3) * 16;
    const f32x4* s4 = (const f32x4*)(W1 + (size_t)(N0 + nr) * 1024 + K0 + kc);
    f32x4 f0 = s4[0], f1 = s4[1], f2 = s4[2], f3 = s4[3];
    ush8 p0, p1;
    #pragma unroll
    for (int i = 0; i < 4; ++i) {
      p0[i]     = f2bf(f0[i]);  p0[4 + i] = f2bf(f1[i]);
      p1[i]     = f2bf(f2[i]);  p1[4 + i] = f2bf(f3[i]);
    }
    *(ush8*)&Tld[nr * 80 + kc]     = p0;
    *(ush8*)&Tld[nr * 80 + kc + 8] = p1;
  }
  __syncthreads();
  {
    const int kr = tid >> 2;           // 0..63  (k row of W1T)
    const int nc = (tid & 3) * 16;
    ush8 p0, p1;
    #pragma unroll
    for (int i = 0; i < 8; ++i) {
      p0[i] = Tld[(nc + i) * 80 + kr];
      p1[i] = Tld[(nc + 8 + i) * 80 + kr];
    }
    unsigned short* d = W1Tb + (size_t)(K0 + kr) * 3072 + N0 + nc;
    *(ush8*)&d[0] = p0;
    *(ush8*)&d[8] = p1;
  }
}

// ---------------------------------------------------------------------------
// k1: y = xb @ W1Tb + b1, de-interleaved to q/kT/v (bf16).
// Block = 256 thr (8 waves), tile 128x128, K-step 32, double-buffered async.
// ---------------------------------------------------------------------------
__global__ __launch_bounds__(256) void qkv_gemm_kernel(
    const unsigned short* __restrict__ xb, const unsigned short* __restrict__ W1Tb,
    const float* __restrict__ b1,
    unsigned short* __restrict__ qb, unsigned short* __restrict__ kTb,
    unsigned short* __restrict__ vb)
{
  const int D = 1024;
  __shared__ alignas(16) unsigned short As[2][128 * 40];   // [m][k], padded
  __shared__ alignas(16) unsigned short Bs[2][32 * 136];   // [k][n], padded

  const int tid  = threadIdx.x;
  const int lane = tid & 31;
  const int wave = tid >> 5;          // 0..7
  const int wm   = wave & 3;          // M quarter (32 rows)
  const int wn   = wave >> 2;         // N half   (64 cols)
  const int hl   = lane >> 4;
  const int nl   = lane & 15;
  const int Row0 = blockIdx.y * 128;
  const int Col0 = blockIdx.x * 128;

  auto issue = [&](int buf, int k0) {
    // A tile 128x32: thread -> row tid/2, 16-half chunk (tid&1)
    {
      const int mr = tid >> 1, ks = (tid & 1) * 16;
      const unsigned short* g = xb + (size_t)(Row0 + mr) * D + k0 + ks;
      unsigned short* l = &As[buf][mr * 40 + ks];
      cp16(g, l); cp16(g + 8, l + 8);
    }
    // B tile 32x128: thread -> row tid/8, 16-half chunk (tid&7)
    {
      const int kr = tid >> 3, nc = (tid & 7) * 16;
      const unsigned short* g = W1Tb + (size_t)(k0 + kr) * 3072 + Col0 + nc;
      unsigned short* l = &Bs[buf][kr * 136 + nc];
      cp16(g, l); cp16(g + 8, l + 8);
    }
  };

  v8f acc[2][4];
  #pragma unroll
  for (int i = 0; i < 2; ++i)
    #pragma unroll
    for (int j = 0; j < 4; ++j)
      #pragma unroll
      for (int e = 0; e < 8; ++e) acc[i][j][e] = 0.0f;

  issue(0, 0);
  const int NSTEP = D / 32;            // 32
  for (int i = 0; i < NSTEP; ++i) {
    const int cur = i & 1;
    if (i + 1 < NSTEP) { issue(cur ^ 1, (i + 1) * 32); wait_async_grp(); }
    else               { wait_async_all(); }
    __syncthreads();

    v16bf a[2], b[4];
    #pragma unroll
    for (int mt = 0; mt < 2; ++mt) {
      const unsigned short* ar = &As[cur][(wm * 32 + mt * 16 + nl) * 40];
      a[mt] = mk16(*(const ush8*)&ar[hl * 8], *(const ush8*)&ar[16 + hl * 8]);
    }
    #pragma unroll
    for (int nt = 0; nt < 4; ++nt) {
      const unsigned short* br = &Bs[cur][wn * 64 + nt * 16 + nl];
      ush16 ub;
      #pragma unroll
      for (int s = 0; s < 16; ++s) ub[s] = br[((s & 7) + 8 * hl + 16 * (s >> 3)) * 136];
      b[nt] = __builtin_bit_cast(v16bf, ub);
    }
    #pragma unroll
    for (int mt = 0; mt < 2; ++mt)
      #pragma unroll
      for (int nt = 0; nt < 4; ++nt)
        acc[mt][nt] = WMMA_BF16(a[mt], b[nt], acc[mt][nt]);

    __syncthreads();
  }

  // ---- epilogue: bias add, de-interleave (n -> head h, chan ch, s=q/k/v) ----
  #pragma unroll
  for (int nt = 0; nt < 4; ++nt) {
    const int n   = Col0 + wn * 64 + nt * 16 + nl;
    const int h   = n / 192;
    const int rem = n - h * 192;
    const int ch  = rem / 3;
    const int s   = rem - ch * 3;
    const float bias = b1[n];
    #pragma unroll
    for (int mt = 0; mt < 2; ++mt) {
      #pragma unroll
      for (int r = 0; r < 8; ++r) {
        const int m  = Row0 + wm * 32 + mt * 16 + r + 8 * hl;
        const int bb = m >> 11;
        const int t  = m & 2047;
        const unsigned short val = f2bf(acc[mt][nt][r] + bias);
        if (s == 0)
          qb [((size_t)(bb * 16 + h) * 2048 + t) * 64 + ch] = val;
        else if (s == 1)
          kTb[((size_t)(bb * 16 + h) * 64 + ch) * 2048 + t] = val;
        else
          vb [((size_t)(bb * 16 + h) * 2048 + t) * 64 + ch] = val;
      }
    }
  }
}

// ---------------------------------------------------------------------------
// k2: flash attention.  grid = (T/64, B*H), block = 128 (4 waves).
// K/V staged to LDS once per WG (async, double buffered); each wave: 16 rows.
// ---------------------------------------------------------------------------
__global__ __launch_bounds__(128) void flash_attn_kernel(
    const unsigned short* __restrict__ qb, const unsigned short* __restrict__ kTb,
    const unsigned short* __restrict__ vb, float* __restrict__ out)
{
  const int T = 2048, dh = 64;
  __shared__ alignas(16) unsigned short Ktl[2][64 * 40];  // [d][32 keys], padded
  __shared__ alignas(16) unsigned short Vtl[2][32 * 80];  // [key][64 ch], padded
  __shared__ alignas(16) unsigned short Pld[4][16 * 40];  // per-wave P, padded

  const int tid  = threadIdx.x;
  const int lane = tid & 31;
  const int wave = tid >> 5;          // 0..3
  const int hl   = lane >> 4;
  const int nl   = lane & 15;
  const int bh   = blockIdx.y;        // b*16 + h
  const int q0   = blockIdx.x * 64 + wave * 16;

  const unsigned short* qp = qb  + (size_t)bh * T * dh;
  const unsigned short* kp = kTb + (size_t)bh * dh * T;
  const unsigned short* vp = vb  + (size_t)bh * T * dh;

  auto issueKV = [&](int buf, int k0) {
    // K^T tile 64(d) x 32(keys): thread -> d = tid/2, chunk (tid&1)
    {
      const int dr = tid >> 1, c2 = (tid & 1) * 16;
      const unsigned short* g = kp + (size_t)dr * T + k0 + c2;
      unsigned short* l = &Ktl[buf][dr * 40 + c2];
      cp16(g, l); cp16(g + 8, l + 8);
    }
    // V tile 32(key) x 64(ch): thread -> key = tid/4, chunk (tid&3)
    {
      const int kr = tid >> 2, cv = (tid & 3) * 16;
      const unsigned short* g = vp + (size_t)(k0 + kr) * dh + cv;
      unsigned short* l = &Vtl[buf][kr * 80 + cv];
      cp16(g, l); cp16(g + 8, l + 8);
    }
  };

  // Q in A-matrix layout, two d-chunks of 32
  v16bf aQ[2];
  #pragma unroll
  for (int j = 0; j < 2; ++j) {
    const unsigned short* r = qp + (size_t)(q0 + nl) * dh + j * 32;
    aQ[j] = mk16(*(const ush8*)&r[hl * 8], *(const ush8*)&r[16 + hl * 8]);
  }

  v8f O[4];
  #pragma unroll
  for (int nt = 0; nt < 4; ++nt)
    #pragma unroll
    for (int e = 0; e < 8; ++e) O[nt][e] = 0.0f;
  float mrow[8], lrow[8];
  #pragma unroll
  for (int r = 0; r < 8; ++r) { mrow[r] = -3.0e38f; lrow[r] = 0.0f; }

  issueKV(0, 0);
  const int NBLK = T / 32;            // 64
  for (int i = 0; i < NBLK; ++i) {
    const int cur = i & 1;
    if (i + 1 < NBLK) { issueKV(cur ^ 1, (i + 1) * 32); wait_async_grp(); }
    else              { wait_async_all(); }
    __syncthreads();

    // ---- S = Q K^T for two 16-key sub-tiles ----
    v8f S[2];
    #pragma unroll
    for (int kt = 0; kt < 2; ++kt) {
      v8f z;
      #pragma unroll
      for (int e = 0; e < 8; ++e) z[e] = 0.0f;
      #pragma unroll
      for (int j = 0; j < 2; ++j) {
        const unsigned short* r = &Ktl[cur][(j * 32) * 40 + kt * 16 + nl];
        ush16 uk;
        #pragma unroll
        for (int s = 0; s < 16; ++s) uk[s] = r[((s & 7) + 8 * hl + 16 * (s >> 3)) * 40];
        v16bf bK = __builtin_bit_cast(v16bf, uk);
        z = WMMA_BF16(aQ[j], bK, z);
      }
      S[kt] = z;
    }

    // ---- online softmax over 16x32 tile; emit P (bf16) into LDS ----
    unsigned short* pw = &Pld[wave][0];
    #pragma unroll
    for (int r = 0; r < 8; ++r) {
      float s0 = S[0][r] * 0.125f;     // 1/sqrt(dh)
      float s1 = S[1][r] * 0.125f;
      float tmax = fmaxf(s0, s1);
      tmax = fmaxf(tmax, __shfl_xor(tmax, 1, 32));
      tmax = fmaxf(tmax, __shfl_xor(tmax, 2, 32));
      tmax = fmaxf(tmax, __shfl_xor(tmax, 4, 32));
      tmax = fmaxf(tmax, __shfl_xor(tmax, 8, 32));  // within 16-lane half
      const float mnew = fmaxf(mrow[r], tmax);
      const float sc   = __expf(mrow[r] - mnew);
      const float p0   = __expf(s0 - mnew);
      const float p1   = __expf(s1 - mnew);
      float ts = p0 + p1;
      ts += __shfl_xor(ts, 1, 32);
      ts += __shfl_xor(ts, 2, 32);
      ts += __shfl_xor(ts, 4, 32);
      ts += __shfl_xor(ts, 8, 32);
      lrow[r] = lrow[r] * sc + ts;
      mrow[r] = mnew;
      #pragma unroll
      for (int nt = 0; nt < 4; ++nt) O[nt][r] *= sc;
      const int row = r + 8 * hl;
      pw[row * 40 + nl]      = f2bf(p0);
      pw[row * 40 + 16 + nl] = f2bf(p1);
    }

    // ---- reload P in A-matrix layout (per-wave private LDS region) ----
    const unsigned short* pr = &Pld[wave][nl * 40];
    v16bf aP = mk16(*(const ush8*)&pr[hl * 8], *(const ush8*)&pr[16 + hl * 8]);

    // ---- O += P V ----
    #pragma unroll
    for (int nt = 0; nt < 4; ++nt) {
      const unsigned short* r = &Vtl[cur][nt * 16 + nl];
      ush16 uv;
      #pragma unroll
      for (int s = 0; s < 16; ++s) uv[s] = r[((s & 7) + 8 * hl + 16 * (s >> 3)) * 80];
      v16bf bV = __builtin_bit_cast(v16bf, uv);
      O[nt] = WMMA_BF16(aP, bV, O[nt]);
    }

    __syncthreads();
  }

  // ---- normalize and write out[b, t, h*64 + c] ----
  const int b = bh >> 4, h = bh & 15;
  #pragma unroll
  for (int r = 0; r < 8; ++r) {
    const float inv = 1.0f / lrow[r];
    const int t = q0 + r + 8 * hl;
    float* o = out + ((size_t)(b * T + t)) * 1024 + h * 64;
    #pragma unroll
    for (int nt = 0; nt < 4; ++nt)
      o[nt * 16 + nl] = O[nt][r] * inv;
  }
}

// ---------------------------------------------------------------------------
extern "C" void kernel_launch(void* const* d_in, const int* in_sizes, int n_in,
                              void* d_out, int out_size, void* d_ws, size_t ws_size,
                              hipStream_t stream) {
  (void)in_sizes; (void)n_in; (void)out_size; (void)ws_size;
  const float* x  = (const float*)d_in[0];
  const float* W1 = (const float*)d_in[1];
  const float* b1 = (const float*)d_in[2];
  // d_in[3]=W2, d_in[4]=b2: dead in the reference (fc2 output is discarded).

  float* out = (float*)d_out;
  unsigned short* ws   = (unsigned short*)d_ws;
  const size_t XB   = (size_t)4096 * 1024;        // 4M halves
  const size_t W1TB = (size_t)1024 * 3072;        // 3M halves
  const size_t QKV  = (size_t)2 * 16 * 2048 * 64; // 4M halves each
  unsigned short* xb   = ws;
  unsigned short* w1tb = xb + XB;
  unsigned short* qb   = w1tb + W1TB;
  unsigned short* kTb  = qb + QKV;
  unsigned short* vb   = kTb + QKV;

  convert_x_kernel<<<dim3(1024), 256, 0, stream>>>(x, xb);
  convert_w1t_kernel<<<dim3(16, 48), 256, 0, stream>>>(W1, w1tb);

  dim3 g1(3072 / 128, 4096 / 128);    // (N tiles, M tiles)
  qkv_gemm_kernel<<<g1, 256, 0, stream>>>(xb, w1tb, b1, qb, kTb, vb);

  dim3 g2(2048 / 64, 2 * 16);         // (query tiles, B*H)
  flash_attn_kernel<<<g2, 128, 0, stream>>>(qb, kTb, vb, out);
}
